// GILayer_80822694576665
// MI455X (gfx1250) — compile-verified, matrix operand order checked by reading, not run on hardware
//
#include <hip/hip_runtime.h>

// ---------------------------------------------------------------------------
// GILayer on MI455X (gfx1250): S=16, N=4096, M=128, D=128, O=128
//   Big einsums -> v_wmma_f32_16x16x32_bf16 (bf16 in, f32 acc)
//   Cholesky / tri-inverse chain -> f32 in LDS (whole 128x128 problem resident)
// ---------------------------------------------------------------------------

#define S_ 16
#define N_ 4096
#define M_ 128
#define D_ 128
#define O_ 128

typedef __bf16 bf16_t;
typedef __attribute__((ext_vector_type(16))) __bf16 v16bf;
typedef __attribute__((ext_vector_type(8)))  float  v8f;

// Load one 16x32 bf16 WMMA operand fragment from a row-major [128 x 128]
// LDS tile (stride 128). Works for both A (rows) and B (cols, when the tile
// is stored transposed so K is contiguous). Per the CDNA5 ISA layout, lane L
// holds row (L&15); VGPRs 0..3 carry K = k0..k0+7 and VGPRs 4..7 carry
// K = k0+16..k0+23 with k0 = 32*chunk + 8*(L>>4) -> two contiguous 8-elem runs.
__device__ __forceinline__ v16bf load_tile_frag(const bf16_t* base, int tile16,
                                                int kchunk, int lane) {
  const bf16_t* rowp = base + (tile16 * 16 + (lane & 15)) * 128;
  const int k0 = kchunk * 32 + ((lane >> 4) << 3);
  union { v16bf v; bf16_t e[16]; } u;
#pragma unroll
  for (int t = 0; t < 8; ++t) {
    u.e[t]     = rowp[k0 + t];
    u.e[8 + t] = rowp[k0 + 16 + t];
  }
  return u.v;
}

// In-place lower Cholesky of a 128x128 f32 matrix in LDS. 256 threads.
__device__ __forceinline__ void chol128_lds(float* A, int tid) {
  for (int k = 0; k < 128; ++k) {
    if (tid == 0) A[k * 128 + k] = sqrtf(A[k * 128 + k]);
    __syncthreads();
    const float dk = A[k * 128 + k];
    for (int i = k + 1 + tid; i < 128; i += 256) A[i * 128 + k] /= dk;
    __syncthreads();
    for (int i = k + 1 + tid; i < 128; i += 256) {
      const float lik = A[i * 128 + k];
      for (int j = k + 1; j <= i; ++j)
        A[i * 128 + j] = fmaf(-lik, A[j * 128 + k], A[i * 128 + j]);
    }
    __syncthreads();
  }
}

__device__ __forceinline__ float block_reduce256(float v, float* red, int tid) {
  red[tid] = v;
  __syncthreads();
  for (int off = 128; off > 0; off >>= 1) {
    if (tid < off) red[tid] += red[tid + off];
    __syncthreads();
  }
  const float r = red[0];
  __syncthreads();
  return r;
}

__global__ void gil_zero_kl(float* kl) {
  if (threadIdx.x < S_) kl[threadIdx.x] = 0.0f;
}

// One block per (s,o). 256 threads = 8 waves. Dynamic LDS layout (200704 B):
//   T   bf16[128][128]  @0       : phiU^T               (A operands)
//   Tw  bf16[128][128]  @32768   : (Lp*phiU)^T          (B operands)
//   A   f32 [128][128]  @65536   : Gram / chol / cov / Lcov (in place)
//   Li  f32 [128][128]  @131072  : L^{-1}
//   small vectors + reduce scratch @196608
__global__ void __launch_bounds__(256) gil_batch_kernel(
    const float* __restrict__ U, const float* __restrict__ pmu,
    const float* __restrict__ plogprec, const float* __restrict__ eps,
    float* __restrict__ Uout, float* __restrict__ klout,
    bf16_t* __restrict__ wbf) {
  extern __shared__ char smem[];
  bf16_t* T    = (bf16_t*)smem;
  bf16_t* Tw   = (bf16_t*)(smem + 32768);
  float*  A    = (float*)(smem + 65536);
  float*  Li   = (float*)(smem + 131072);
  float*  lpS  = (float*)(smem + 196608);
  float*  lmS  = lpS + 128;
  float*  vS   = lpS + 256;
  float*  qmuS = lpS + 384;
  float*  wS   = lpS + 512;
  float*  epsS = lpS + 640;
  float*  red  = lpS + 768;  // 256 floats

  const int tid  = threadIdx.x;
  const int lane = tid & 31;
  const int wave = tid >> 5;
  const int o = blockIdx.x;
  const int s = blockIdx.y;

  if (tid < 128) {
    const float lp = expf(plogprec[o * M_ + tid]);
    lpS[tid]  = lp;
    lmS[tid]  = lp * pmu[o * M_ + tid];
    epsS[tid] = eps[((size_t)(s * O_ + o)) * D_ + tid];
  }
  __syncthreads();

  // Stage transposed bf16 phiU and Lp-weighted phiU.
  const float* Us = U + (size_t)s * M_ * D_;
  for (int idx = tid; idx < 128 * 128; idx += 256) {
    const int m = idx >> 7, d = idx & 127;
    const float p = fmaxf(Us[idx], 0.0f);
    T[d * 128 + m]  = (bf16_t)p;
    Tw[d * 128 + m] = (bf16_t)(p * lpS[m]);
  }
  __syncthreads();

  // A = phiU^T diag(Lp) phiU  (128x128x128 GEMM via WMMA, 8x8 tiles).
  {
    const int i = wave;
#pragma unroll
    for (int j = 0; j < 8; ++j) {
      v8f acc = {};
#pragma unroll
      for (int c = 0; c < 4; ++c) {
        v16bf a = load_tile_frag(T, i, c, lane);
        v16bf b = load_tile_frag(Tw, j, c, lane);
        acc = __builtin_amdgcn_wmma_f32_16x16x32_bf16(
            false, a, false, b, (short)0, acc, false, false);
      }
      const int col   = j * 16 + (lane & 15);
      const int rbase = i * 16 + ((lane >> 4) << 3);
#pragma unroll
      for (int r = 0; r < 8; ++r) A[(rbase + r) * 128 + col] = acc[r];
    }
  }
  __syncthreads();

  // q_prec = A + I ; UTLv = phiU^T (Lp*mu)
  if (tid < 128) {
    A[tid * 128 + tid] += 1.0f;
    float v = 0.0f;
    for (int m = 0; m < 128; ++m) v = fmaf((float)T[tid * 128 + m], lmS[m], v);
    vS[tid] = v;
  }
  __syncthreads();

  chol128_lds(A, tid);  // A = L (lower)

  // Li = L^{-1} by forward substitution, one column per thread.
  if (tid < 128) {
    const int c = tid;
    for (int i = 0; i < c; ++i) Li[i * 128 + c] = 0.0f;
    for (int i = c; i < 128; ++i) {
      float ssum = (i == c) ? 1.0f : 0.0f;
      for (int j = c; j < i; ++j)
        ssum = fmaf(-A[i * 128 + j], Li[j * 128 + c], ssum);
      Li[i * 128 + c] = ssum / A[i * 128 + i];
    }
  }
  __syncthreads();

  // q_cov = Li^T Li  (overwrite A; chol factor no longer needed).
  for (int idx = tid; idx < 128 * 128; idx += 256) {
    const int i = idx >> 7, j = idx & 127;
    const int k0 = i > j ? i : j;
    float sacc = 0.0f;
    for (int k = k0; k < 128; ++k)
      sacc = fmaf(Li[k * 128 + i], Li[k * 128 + j], sacc);
    A[idx] = sacc;
  }
  __syncthreads();

  const float trc = block_reduce256(tid < 128 ? A[tid * 128 + tid] : 0.0f, red, tid);
  if (tid < 128) {
    float q = 0.0f;
    for (int e = 0; e < 128; ++e) q = fmaf(A[tid * 128 + e], vS[e], q);
    qmuS[tid] = q;
  }
  __syncthreads();
  const float musq =
      block_reduce256(tid < 128 ? qmuS[tid] * qmuS[tid] : 0.0f, red, tid);

  chol128_lds(A, tid);  // A = Lcov = chol(q_cov)

  const float logdet = block_reduce256(
      tid < 128 ? 2.0f * logf(A[tid * 128 + tid]) : 0.0f, red, tid);

  // w = q_mu + Lcov @ eps
  if (tid < 128) {
    float acc = qmuS[tid];
    for (int e = 0; e <= tid; ++e) acc = fmaf(A[tid * 128 + e], epsS[e], acc);
    wS[tid] = acc;
  }
  __syncthreads();

  if (tid == 0) {
    const float kl = 0.5f * (trc + musq - (float)D_ - logdet);
    atomicAdd(&klout[s], kl);
  }

  // U_out[s, m, o] = sum_d phiU[m,d] * w[d]
  if (tid < 128) {
    const int m = tid;
    float acc = 0.0f;
    for (int d = 0; d < 128; ++d)
      acc = fmaf((float)T[d * 128 + m], wS[d], acc);
    Uout[((size_t)s * M_ + m) * O_ + o] = acc;
  }

  // bf16 w laid out [S][O][D] so F_out's B fragments are K-contiguous.
  if (tid < 128) wbf[((size_t)(s * O_ + o)) * D_ + tid] = (bf16_t)wS[tid];
}

// One block per (s, 128-row tile of N). F_out = relu(F) @ w^T via WMMA.
__global__ void __launch_bounds__(256) gil_fout_kernel(
    const float* __restrict__ F, const bf16_t* __restrict__ wbf,
    float* __restrict__ Fout) {
  extern __shared__ char smem[];
  bf16_t* Fr = (bf16_t*)smem;             // [128 n][128 d]
  bf16_t* Wt = (bf16_t*)(smem + 32768);   // [128 o][128 d]

  const int tid  = threadIdx.x;
  const int lane = tid & 31;
  const int wave = tid >> 5;
  const int nb = blockIdx.x;
  const int s  = blockIdx.y;

  // Stage w matrix (dword copies) and relu(F) tile (with prefetch ahead).
  const unsigned* wsrc = (const unsigned*)(wbf + (size_t)s * O_ * D_);
  unsigned* wdst = (unsigned*)Wt;
  for (int idx = tid; idx < 128 * 128 / 2; idx += 256) wdst[idx] = wsrc[idx];

  const float* Fs = F + ((size_t)s * N_ + (size_t)nb * 128) * D_;
  for (int idx = tid; idx < 128 * 128; idx += 256) {
    if (idx + 2048 < 128 * 128) __builtin_prefetch(Fs + idx + 2048, 0, 1);
    Fr[idx] = (bf16_t)fmaxf(Fs[idx], 0.0f);
  }
  __syncthreads();

  const int i = wave;
#pragma unroll
  for (int j = 0; j < 8; ++j) {
    v8f acc = {};
#pragma unroll
    for (int c = 0; c < 4; ++c) {
      v16bf a = load_tile_frag(Fr, i, c, lane);
      v16bf b = load_tile_frag(Wt, j, c, lane);
      acc = __builtin_amdgcn_wmma_f32_16x16x32_bf16(
          false, a, false, b, (short)0, acc, false, false);
    }
    const int o    = j * 16 + (lane & 15);
    const int nloc = i * 16 + ((lane >> 4) << 3);
    float* outp = Fout + ((size_t)s * N_ + (size_t)nb * 128 + nloc) * O_ + o;
#pragma unroll
    for (int r = 0; r < 8; ++r) outp[(size_t)r * O_] = acc[r];
  }
}

extern "C" void kernel_launch(void* const* d_in, const int* in_sizes, int n_in,
                              void* d_out, int out_size, void* d_ws, size_t ws_size,
                              hipStream_t stream) {
  const float* F   = (const float*)d_in[0];
  const float* U   = (const float*)d_in[1];
  const float* pmu = (const float*)d_in[2];
  const float* plp = (const float*)d_in[3];
  const float* eps = (const float*)d_in[4];

  float* Fout = (float*)d_out;                          // [S,N,O]
  float* Uout = Fout + (size_t)S_ * N_ * O_;            // [S,M,O]
  float* kl   = Uout + (size_t)S_ * M_ * O_;            // [S]
  bf16_t* wbf = (bf16_t*)d_ws;                          // [S,O,D] bf16

  gil_zero_kl<<<1, 32, 0, stream>>>(kl);

  // 200704 B dynamic LDS: 2x32K bf16 staging + 2x64K f32 + 4K vectors.
  const size_t lds1 = 196608 + 1024 * sizeof(float);
  gil_batch_kernel<<<dim3(O_, S_), 256, lds1, stream>>>(U, pmu, plp, eps,
                                                        Uout, kl, wbf);

  // 64 KB dynamic LDS: bf16 F tile + bf16 w matrix.
  gil_fout_kernel<<<dim3(N_ / 128, S_), 256, 65536, stream>>>(F, wbf, Fout);
}
